// CRF2oDependency_4733053960798
// MI455X (gfx1250) — compile-verified, matrix operand order checked by reading, not run on hardware
//
#include <hip/hip_runtime.h>
#include <hip/hip_bf16.h>
#include <stdint.h>

// ============================================================================
// CRF 2nd-order dependency inside algorithm on MI455X (gfx1250).
//
//  * One workgroup (1024 threads = 32 wave32) per batch element; DP state
//    (s_i, s_s, s_c : 3 x 128x128 f32 = 192 KB) lives in LDS (<=320KB/WGP).
//  * 127 sequential width steps, 2 workgroup barriers per step (race-free by
//    diagonal-offset analysis: each phase reads offsets < w, writes offset w).
//  * Two-pass logsumexp (max pass, then exp pass) with a SINGLE v8f WMMA
//    accumulator — deliberately recomputing terms instead of caching them in
//    registers.
//  * exp(x - rowmax) values are packed into the ISA 16x32 f16 A-tile layout
//    and row-summed with v_wmma_f32_16x16x32_f16 against an all-ones B:
//    D[m,n] = sum_k A[m,k] -> 16 rows x 32 terms per WMMA.
//  * Branchless terms (clamped addresses + v_cndmask); reduction type is a
//    template parameter dispatched on a scalar task id (no exec divergence).
//  * Extraction: 7-cndmask dynamic element select + ONE ds_bpermute routes
//    row sums to lanes 0..15, which each do ONE log and one coalesced store.
// ============================================================================

#define SDIM 128
#define NEGV  (-1e30f)
#define NEG_TH (-1e29f)

typedef __attribute__((ext_vector_type(16))) _Float16 v16h;
typedef __attribute__((ext_vector_type(8)))  float    v8f;

// Branchless logsumexp term. kc pre-clamped to [0, nk-1]; all addresses are
// in-bounds by construction; invalid terms masked to NEGV via selects.
// TYPE: 0=il (left incomplete), 1=ir (right incomplete), 2=sibling span,
//       3=cl (left complete),   4=cr (right complete)
template <int TYPE>
__device__ __forceinline__ float dp_term_bl(int kc, int j, int w, bool valid,
                                            const float* __restrict__ sI,
                                            const float* __restrict__ sS,
                                            const float* __restrict__ sC,
                                            const float* __restrict__ sib_b) {
  const int kw = kc + w;
  const int jc = (j < w) ? j : (w - 1);          // clamp for addressing
  float x;
  if constexpr (TYPE == 0) {                     // I(k+w -> k)
    const int wm2 = (w >= 2) ? (w - 2) : 0;
    const int jr  = (jc < wm2) ? jc : wm2;       // regular-term j in [0, w-2]
    const int t   = kc + jr + 1;
    const float reg = sI[kw * SDIM + t] + sS[t * SDIM + kc]
                    + sib_b[(kc * SDIM + kw) * SDIM + t];   // s_sib[b,k,kw,t]
    // special j==w-1 term: sC[kw,kw] + sC[k,kw-1]; the diagonal is always 0.
    const float spc = (kc == 0) ? 0.0f : sC[kc * SDIM + (kw - 1)];
    x = (j == w - 1) ? spc : reg;
  } else if constexpr (TYPE == 1) {              // I(k -> k+w)
    const int jr = (jc >= 1) ? jc : 1;           // regular-term j in [1, w-1]
    const int t  = kc + jr;
    const float reg = sI[kc * SDIM + t] + sS[t * SDIM + kw]
                    + sib_b[(kw * SDIM + kc) * SDIM + t];   // s_sib[b,kw,k,t]
    const float spc = sC[kw * SDIM + (kc + 1)];  // + sC[k,k] == 0
    x = (j == 0) ? spc : ((kc == 0) ? NEGV : reg);
  } else if constexpr (TYPE == 2) {              // S(k, k+w)
    const int t = kc + jc;
    x = sC[kc * SDIM + t] + sC[kw * SDIM + (t + 1)];
  } else if constexpr (TYPE == 3) {              // C(k+w -> k)
    const int t = kc + jc;
    x = sC[t * SDIM + kc] + sI[kw * SDIM + t];
  } else {                                       // C(k -> k+w)
    const int t = kc + jc + 1;
    x = sI[kc * SDIM + t] + sC[t * SDIM + kw];
  }
  return valid ? x : NEGV;
}

// Select element (lane & 7) from a v8f with a 3-level cndmask tree.
__device__ __forceinline__ float dynsel8(v8f v, int lane) {
  const bool l0 = (lane & 1) != 0;
  const bool l1 = (lane & 2) != 0;
  const bool l2 = (lane & 4) != 0;
  const float a0 = l0 ? v[1] : v[0];
  const float a1 = l0 ? v[3] : v[2];
  const float a2 = l0 ? v[5] : v[4];
  const float a3 = l0 ? v[7] : v[6];
  const float b0 = l1 ? a1 : a0;
  const float b1 = l1 ? a3 : a2;
  return l2 ? b1 : b0;
}

// One wave reduces a tile of 16 consecutive k-rows over j in [0, w) with
// logsumexp; the exp-sum runs on the matrix pipe via WMMA.
template <int TYPE>
__device__ __forceinline__ void dp_reduce_tile(int tile, int w, int nk, int lenB,
                                               float* __restrict__ sI,
                                               float* __restrict__ sS,
                                               float* __restrict__ sC,
                                               const float* __restrict__ sib_b,
                                               const float* __restrict__ arc_b,
                                               int lane) {
  const int hi = lane >> 4;       // which K-half of the A tile this lane feeds
  const int m  = lane & 15;       // matrix row M handled by this lane
  const int k  = (tile << 4) + m;
  const bool rowv = (k < nk);
  const int kc = rowv ? k : (nk - 1);            // clamped row for addressing
  const int nch = (w + 31) >> 5;                 // 32 K-terms per WMMA chunk

  // ---- pass 1: per-row max (lane l and l+16 both hold row m = l&15) ----
  float mx = NEGV;
  for (int ch = 0; ch < nch; ++ch) {
    const int jb = ch << 5;
    #pragma unroll
    for (int e = 0; e < 16; ++e) {
      // ISA 16-bit A-matrix 16x32 layout: lanes 0-15 hold K 0-7,16-23;
      // lanes 16-31 hold K 8-15,24-31 (element e -> VGPR e/2, lo/hi half).
      const int K = ((e < 8) ? e : (e + 8)) + (hi << 3);
      const int j = jb + K;
      mx = fmaxf(mx, dp_term_bl<TYPE>(kc, j, w, rowv && (j < w),
                                      sI, sS, sC, sib_b));
    }
  }
  mx = fmaxf(mx, __shfl_xor(mx, 16, 32));
  const float rowmax = mx;

  // ---- pass 2: exp(x - rowmax) -> f16 A tile, row-sum via WMMA vs ones ----
  v16h ones;
  #pragma unroll
  for (int e = 0; e < 16; ++e) ones[e] = (_Float16)1.0f;

  v8f c = {};
  for (int ch = 0; ch < nch; ++ch) {
    const int jb = ch << 5;
    v16h a;
    #pragma unroll
    for (int e = 0; e < 16; ++e) {
      const int K = ((e < 8) ? e : (e + 8)) + (hi << 3);
      const int j = jb + K;
      const float x  = dp_term_bl<TYPE>(kc, j, w, rowv && (j < w),
                                        sI, sS, sC, sib_b);
      const float ev = (x > NEG_TH) ? __expf(x - rowmax) : 0.0f;
      a[e] = (_Float16)ev;
    }
    // D = A x ones + C : every column of row m holds sum_j exp(x[m,j]-max[m])
    c = __builtin_amdgcn_wmma_f32_16x16x32_f16(false, a, false, ones,
                                               (short)0, c, false, false);
  }

  // ---- extraction: route row sums to lanes 0..15 with one bpermute ----
  // Row r's sum lives in element (r&7) of lanes {r<8: 0..15, r>=8: 16..31}.
  const int   src = (lane & 8) ? (lane + 8) : lane;
  const float tot = __shfl(dynsel8(c, lane), src, 32);

  // lanes 0..15 hold (tot, rowmax) for row == lane; one log + store each.
  const float res0 = (rowmax > NEG_TH) ? (__logf(tot) + rowmax) : NEGV;
  if (lane < 16) {
    const int kk = (tile << 4) + lane;
    if (kk < nk) {
      const int kw = kk + w;
      float res = res0;
      if constexpr (TYPE == 0) {
        sI[kw * SDIM + kk] = res + arc_b[kk * SDIM + kw];   // + A[kw,kk]
      } else if constexpr (TYPE == 1) {
        sI[kk * SDIM + kw] = res + arc_b[kw * SDIM + kk];   // + A[kk,kw]
      } else if constexpr (TYPE == 2) {
        sS[kw * SDIM + kk] = res;
        sS[kk * SDIM + kw] = res;
      } else if constexpr (TYPE == 3) {
        sC[kw * SDIM + kk] = res;
      } else {
        // Fold the single-root constraint into the C(0 -> w) write.
        if (kk == 0 && w != lenB) res = NEGV;
        sC[kk * SDIM + kw] = res;
      }
    }
  }
}

__global__ __launch_bounds__(1024, 1) void crf2o_dp_kernel(
    const float* __restrict__ s_arc, const float* __restrict__ s_sib,
    const unsigned char* __restrict__ mask, const int* __restrict__ arcs,
    const int* __restrict__ sibs, float* __restrict__ ws) {
  const int b        = blockIdx.x;
  const int tid      = threadIdx.x;
  const int nthreads = blockDim.x;

  extern __shared__ float lds[];
  float* sI  = lds;                    // [S*S] incomplete spans  [head*S+dep]
  float* sS  = lds + SDIM * SDIM;      // [S*S] sibling spans
  float* sC  = lds + 2 * SDIM * SDIM;  // [S*S] complete spans
  float* red = lds + 3 * SDIM * SDIM;  // [1024] reduction scratch

  const float*         arc_b  = s_arc + (size_t)b * SDIM * SDIM;
  const float*         sib_b  = s_sib + (size_t)b * SDIM * SDIM * SDIM;
  const unsigned char* mask_b = mask + b * SDIM;

  // ---------------- gold score for this batch (fixed-order tree reduce) ----
  {
    float acc = 0.0f;
    const int* arcs_b = arcs + b * SDIM;
    for (int i = tid; i < SDIM; i += nthreads)
      if (mask_b[i]) acc += arc_b[i * SDIM + arcs_b[i]];
    const int* sibs_b = sibs + (size_t)b * SDIM * SDIM;
    for (int i = tid; i < SDIM * SDIM; i += nthreads) {
      const int sv = sibs_b[i];
      if (sv > 0) acc += sib_b[(size_t)i * SDIM + sv];
    }
    red[tid] = acc;
    __syncthreads();
    for (int off = nthreads >> 1; off > 0; off >>= 1) {
      if (tid < off) red[tid] += red[tid + off];
      __syncthreads();
    }
    if (tid == 0) ws[16 + b] = red[0];
  }

  int lenB = 0;
  for (int s = 0; s < SDIM; ++s) lenB += (int)mask_b[s];
  lenB = __builtin_amdgcn_readfirstlane(lenB);   // uniform -> SGPR

  // ---------------- init DP state in LDS ----------------
  for (int i = tid; i < SDIM * SDIM; i += nthreads) {
    sI[i] = NEGV; sS[i] = NEGV; sC[i] = NEGV;
  }
  __syncthreads();
  for (int d = tid; d < SDIM; d += nthreads) sC[d * SDIM + d] = 0.0f;
  __syncthreads();

  // Scalarize the wave id so task/type/tile live in SGPRs and the type
  // dispatch below becomes scalar branches (no exec-mask divergence).
  const int wave   = __builtin_amdgcn_readfirstlane(tid >> 5);
  const int lane   = tid & 31;
  const int nwaves = nthreads >> 5;   // 32 waves

  // ---------------- sequential DP over widths ----------------
  for (int w = 1; w < SDIM; ++w) {
    const int nk     = SDIM - w;
    const int ntiles = (nk + 15) >> 4;

    // Phase A: il(0), ir(1), sibling(2) — read offsets < w, write offset w.
    for (int task = wave; task < 3 * ntiles; task += nwaves) {
      const int type = task / ntiles;
      const int tile = task - type * ntiles;
      if (type == 0)
        dp_reduce_tile<0>(tile, w, nk, lenB, sI, sS, sC, sib_b, arc_b, lane);
      else if (type == 1)
        dp_reduce_tile<1>(tile, w, nk, lenB, sI, sS, sC, sib_b, arc_b, lane);
      else
        dp_reduce_tile<2>(tile, w, nk, lenB, sI, sS, sC, sib_b, arc_b, lane);
    }
    __syncthreads();

    // Phase B: cl(3), cr(4) — consume the fresh width-w incomplete spans.
    for (int task = wave; task < 2 * ntiles; task += nwaves) {
      const int type = task / ntiles;
      const int tile = task - type * ntiles;
      if (type == 0)
        dp_reduce_tile<3>(tile, w, nk, lenB, sI, sS, sC, sib_b, arc_b, lane);
      else
        dp_reduce_tile<4>(tile, w, nk, lenB, sI, sS, sC, sib_b, arc_b, lane);
    }
    __syncthreads();
  }

  if (tid == 0) ws[b] = sC[lenB];              // logZ_b = s_c[b, 0, len_b]
}

__global__ void crf2o_combine_kernel(const float* __restrict__ ws,
                                     const unsigned char* __restrict__ mask,
                                     int B, float* __restrict__ out) {
  if (threadIdx.x == 0 && blockIdx.x == 0) {
    float logZ = 0.0f, score = 0.0f;
    for (int b = 0; b < B; ++b) { logZ += ws[b]; score += ws[16 + b]; }
    int msum = 0;
    for (int i = 0; i < B * SDIM; ++i) msum += (int)mask[i];
    out[0] = (logZ - score) / (float)msum;
  }
}

extern "C" void kernel_launch(void* const* d_in, const int* in_sizes, int n_in,
                              void* d_out, int out_size, void* d_ws, size_t ws_size,
                              hipStream_t stream) {
  const float*         s_arc = (const float*)d_in[0];         // [B,S,S] f32
  const float*         s_sib = (const float*)d_in[1];         // [B,S,S,S] f32
  const unsigned char* mask  = (const unsigned char*)d_in[2]; // [B,S] bool
  const int*           arcs  = (const int*)d_in[3];           // [B,S] i32
  const int*           sibs  = (const int*)d_in[4];           // [B,S,S] i32
  float* out = (float*)d_out;
  float* ws  = (float*)d_ws;

  const int B = in_sizes[0] / (SDIM * SDIM);   // 16
  const size_t shmem = (size_t)(3 * SDIM * SDIM + 1024) * sizeof(float); // ~196KB

  // LDS request exceeds the 64KB default cap; raise the limit (CDNA5: 320KB/WGP).
  (void)hipFuncSetAttribute((const void*)crf2o_dp_kernel,
                            hipFuncAttributeMaxDynamicSharedMemorySize,
                            (int)shmem);

  crf2o_dp_kernel<<<B, 1024, shmem, stream>>>(s_arc, s_sib, mask, arcs, sibs, ws);
  crf2o_combine_kernel<<<1, 32, 0, stream>>>(ws, mask, B, out);
}
